// SE3Attention_31963146617216
// MI455X (gfx1250) — matrix operand order repaired and checked by reading, not compiled, not run on hardware
//
#include <hip/hip_runtime.h>
#include <math.h>

#define NS 16
#define NV 8
#define HE 32
#define DD 40            // NS + 3*NV
#define WN 640
#define N_NODES 10000
#define N_EDGES 160000
#define NTILES (N_EDGES / 16)   // 10000 exact
#define C110K 0.57735026918962576451f   // 1/sqrt(3)
#define C111K 0.70710678118654752440f   // 1/sqrt(2)
#define EPSBN 1e-5f
#define BW 2             // waves per block (64 threads)

typedef __attribute__((ext_vector_type(16))) _Float16 v16h;
typedef __attribute__((ext_vector_type(8)))  float    v8f;

__device__ __forceinline__ unsigned int encF(float f) {
  unsigned int u = __float_as_uint(f);
  return (u & 0x80000000u) ? ~u : (u | 0x80000000u);   // order-preserving map
}
__device__ __forceinline__ float decF(unsigned int u) {
  unsigned int v = (u & 0x80000000u) ? (u & 0x7fffffffu) : ~u;
  return __uint_as_float(v);
}

// ---------------------------------------------------------------------------
// Per-node q = irreps_linear(atom_features)
// ---------------------------------------------------------------------------
__global__ __launch_bounds__(256)
void node_q_kernel(const float* __restrict__ atom, const float* __restrict__ Wqs,
                   const float* __restrict__ Wqv, float* __restrict__ q) {
  int n = blockIdx.x * 256 + threadIdx.x;
  if (n >= N_NODES) return;
  const float* x = atom + (size_t)n * DD;
  float s[NS];
#pragma unroll
  for (int i = 0; i < NS; ++i) s[i] = x[i];
#pragma unroll
  for (int o = 0; o < NS; ++o) {
    float acc = 0.f;
#pragma unroll
    for (int i = 0; i < NS; ++i) acc += s[i] * Wqs[i * NS + o];
    q[(size_t)n * DD + o] = acc;
  }
  float v[NV][3];
#pragma unroll
  for (int i = 0; i < NV; ++i)
#pragma unroll
    for (int d = 0; d < 3; ++d) v[i][d] = x[NS + i * 3 + d];
#pragma unroll
  for (int o = 0; o < NV; ++o)
#pragma unroll
    for (int d = 0; d < 3; ++d) {
      float acc = 0.f;
#pragma unroll
      for (int i = 0; i < NV; ++i) acc += v[i][d] * Wqv[i * NV + o];
      q[(size_t)n * DD + NS + o * 3 + d] = acc;
    }
}

// ---------------------------------------------------------------------------
// Fused: MLP (WMMA f16) -> tensor product -> (k-pass: attn+segmax | v-pass: msg)
// One wave handles a 16-edge tile. Static LDS = 61,824 B.
// ---------------------------------------------------------------------------
__global__ __launch_bounds__(BW * 32)
void edge_pass_kernel(const float* __restrict__ atom,
                      const float* __restrict__ ef,
                      const float* __restrict__ esh,
                      const float* __restrict__ w1g,
                      const float* __restrict__ b1g,
                      const float* __restrict__ w2g,
                      const float* __restrict__ b2g,
                      const int*   __restrict__ eidx,
                      const float* __restrict__ q,
                      float* __restrict__ vout,
                      float* __restrict__ attn,
                      unsigned int* __restrict__ mEnc,
                      int isK) {
  // Weight B-fragments in f16, pre-laid-out: element j of lane L holds
  // B[K = (L/16)*16 + j][N = nt*16 + L%16]  (contiguous K order, same as the A fill)
  __shared__ alignas(32) _Float16 wB1[2 * 512];
  __shared__ alignas(32) _Float16 wB2[40 * 512];
  __shared__ float b1L[HE];
  __shared__ float b2L[WN];
  // per-wave tile buffers
  __shared__ float stage[BW][16][33];     // ef tile, then hidden tile
  __shared__ float xsL[BW][16][NS];       // x_s
  __shared__ float xsshL[BW][16][NS];     // x_s * sh_s
  __shared__ float dotL[BW][16][NV];      // C110 * <x_v, sh_v>
  __shared__ float xvL[BW][16][NV][3];    // x_v
  __shared__ float crL[BW][16][NV][3];    // C111 * (x_v x sh_v)
  __shared__ float shsL[BW][16];
  __shared__ float shvL[BW][16][3];
  __shared__ int   srcL[BW][16];

  const int tid = threadIdx.x;
  for (int idx = tid; idx < 2 * 512; idx += BW * 32) {
    int nt = idx >> 9, rem = idx & 511, L = rem >> 4, j = rem & 15;
    wB1[idx] = (_Float16)w1g[((L >> 4) * 16 + j) * HE + nt * 16 + (L & 15)];
  }
  for (int idx = tid; idx < 40 * 512; idx += BW * 32) {
    int nt = idx >> 9, rem = idx & 511, L = rem >> 4, j = rem & 15;
    wB2[idx] = (_Float16)w2g[((L >> 4) * 16 + j) * WN + nt * 16 + (L & 15)];
  }
  for (int i = tid; i < HE; i += BW * 32) b1L[i] = b1g[i];
  for (int i = tid; i < WN; i += BW * 32) b2L[i] = b2g[i];
  __syncthreads();

  const int wave = tid >> 5;
  const int lane = tid & 31;
  const int h    = lane >> 4;   // lane half
  const int nn   = lane & 15;   // N within tile / edge id for gather
  const int gw   = blockIdx.x * BW + wave;
  const int tw   = gridDim.x * BW;
  const int iters = (NTILES + tw - 1) / tw;   // uniform across block

  for (int it = 0; it < iters; ++it) {
    const int t = gw + it * tw;
    const bool active = (t < NTILES);
    const int e0 = t * 16;
    __syncthreads();
    if (active) {
      // stage edge-feature tile [16 edges x 32] (coalesced)
#pragma unroll
      for (int k = 0; k < 16; ++k)
        stage[wave][k][lane] = ef[(size_t)(e0 + k) * HE + lane];
      // per-edge gather: 2 lanes per edge (edge = nn, part = h)
      const int ei  = e0 + nn;
      const int dst = eidx[ei];
      if (h == 0) {
        float s0 = esh[ei * 4];
        shsL[wave][nn] = s0;
        srcL[wave][nn] = eidx[N_EDGES + ei];
#pragma unroll
        for (int i = 0; i < NS; ++i) {
          float s = atom[(size_t)dst * DD + i];
          xsL[wave][nn][i]   = s;
          xsshL[wave][nn][i] = s * s0;
        }
      } else {
        float v0 = esh[ei * 4 + 1], v1 = esh[ei * 4 + 2], v2 = esh[ei * 4 + 3];
        shvL[wave][nn][0] = v0; shvL[wave][nn][1] = v1; shvL[wave][nn][2] = v2;
#pragma unroll
        for (int o = 0; o < NV; ++o) {
          float a0 = atom[(size_t)dst * DD + NS + o * 3 + 0];
          float a1 = atom[(size_t)dst * DD + NS + o * 3 + 1];
          float a2 = atom[(size_t)dst * DD + NS + o * 3 + 2];
          xvL[wave][nn][o][0] = a0; xvL[wave][nn][o][1] = a1; xvL[wave][nn][o][2] = a2;
          dotL[wave][nn][o] = C110K * (a0 * v0 + a1 * v1 + a2 * v2);
          crL[wave][nn][o][0] = C111K * (a1 * v2 - a2 * v1);
          crL[wave][nn][o][1] = C111K * (a2 * v0 - a0 * v2);
          crL[wave][nn][o][2] = C111K * (a0 * v1 - a1 * v0);
        }
      }
    }
    __syncthreads();
    if (active) {
      // ---- layer 1: hidden = relu(ef @ W1 + b1) ----
      v16h a1;
#pragma unroll
      for (int j = 0; j < 16; ++j) {
        int K = h * 16 + j;   // A layout K map
        a1[j] = (_Float16)stage[wave][nn][K];
      }
      float hv[2][8];
#pragma unroll
      for (int nt = 0; nt < 2; ++nt) {
        v8f c;
        float bb = b1L[nt * 16 + nn];
#pragma unroll
        for (int r = 0; r < 8; ++r) c[r] = bb;
        v16h b = *(const v16h*)(wB1 + nt * 512 + lane * 16);
        c = __builtin_amdgcn_wmma_f32_16x16x32_f16(false, a1, false, b, (short)0,
                                                   c, false, false);
#pragma unroll
        for (int r = 0; r < 8; ++r) hv[nt][r] = fmaxf(c[r], 0.0f);
      }
#pragma unroll
      for (int nt = 0; nt < 2; ++nt)
#pragma unroll
        for (int r = 0; r < 8; ++r)
          stage[wave][h * 8 + r][nt * 16 + nn] = hv[nt][r];   // D->A relayout via LDS
    }
    __syncthreads();
    if (active) {
      v16h a2;
#pragma unroll
      for (int j = 0; j < 16; ++j) {
        int K = h * 16 + j;
        a2[j] = (_Float16)stage[wave][nn][K];
      }
      // ---- layer 2 GEMM fused with tensor product ----
      // lane owns weight column n = nt*16+nn for edges m = h*8+r
      float souts[8], vAa[8], vBa[8][3], vCa[8][3];
#pragma unroll
      for (int r = 0; r < 8; ++r) {
        souts[r] = 0.f; vAa[r] = 0.f;
#pragma unroll
        for (int d = 0; d < 3; ++d) { vBa[r][d] = 0.f; vCa[r][d] = 0.f; }
      }
      const int ib = nn >> 3;   // i-parity group for NV-wide regions
      for (int nt = 0; nt < 16; ++nt) {            // w1: i=nt, o=nn
        v8f c; float bb = b2L[nt * 16 + nn];
#pragma unroll
        for (int r = 0; r < 8; ++r) c[r] = bb;
        v16h b = *(const v16h*)(wB2 + nt * 512 + lane * 16);
        c = __builtin_amdgcn_wmma_f32_16x16x32_f16(false, a2, false, b, (short)0, c, false, false);
#pragma unroll
        for (int r = 0; r < 8; ++r) souts[r] += c[r] * xsshL[wave][h * 8 + r][nt];
      }
      for (int nt = 16; nt < 24; ++nt) {           // w2: i=nt-16, o=nn (C110 in dotL)
        v8f c; float bb = b2L[nt * 16 + nn];
#pragma unroll
        for (int r = 0; r < 8; ++r) c[r] = bb;
        v16h b = *(const v16h*)(wB2 + nt * 512 + lane * 16);
        c = __builtin_amdgcn_wmma_f32_16x16x32_f16(false, a2, false, b, (short)0, c, false, false);
#pragma unroll
        for (int r = 0; r < 8; ++r) souts[r] += c[r] * dotL[wave][h * 8 + r][nt - 16];
      }
      for (int nt = 24; nt < 32; ++nt) {           // w3: i=(nt-24)*2+ib, o=nn&7
        v8f c; float bb = b2L[nt * 16 + nn];
#pragma unroll
        for (int r = 0; r < 8; ++r) c[r] = bb;
        v16h b = *(const v16h*)(wB2 + nt * 512 + lane * 16);
        c = __builtin_amdgcn_wmma_f32_16x16x32_f16(false, a2, false, b, (short)0, c, false, false);
        int i = (nt - 24) * 2 + ib;
#pragma unroll
        for (int r = 0; r < 8; ++r) vAa[r] += c[r] * xsL[wave][h * 8 + r][i];
      }
      for (int nt = 32; nt < 36; ++nt) {           // w4
        v8f c; float bb = b2L[nt * 16 + nn];
#pragma unroll
        for (int r = 0; r < 8; ++r) c[r] = bb;
        v16h b = *(const v16h*)(wB2 + nt * 512 + lane * 16);
        c = __builtin_amdgcn_wmma_f32_16x16x32_f16(false, a2, false, b, (short)0, c, false, false);
        int i = (nt - 32) * 2 + ib;
#pragma unroll
        for (int r = 0; r < 8; ++r)
#pragma unroll
          for (int d = 0; d < 3; ++d) vBa[r][d] += c[r] * xvL[wave][h * 8 + r][i][d];
      }
      for (int nt = 36; nt < 40; ++nt) {           // w5 (C111 in crL)
        v8f c; float bb = b2L[nt * 16 + nn];
#pragma unroll
        for (int r = 0; r < 8; ++r) c[r] = bb;
        v16h b = *(const v16h*)(wB2 + nt * 512 + lane * 16);
        c = __builtin_amdgcn_wmma_f32_16x16x32_f16(false, a2, false, b, (short)0, c, false, false);
        int i = (nt - 36) * 2 + ib;
#pragma unroll
        for (int r = 0; r < 8; ++r)
#pragma unroll
          for (int d = 0; d < 3; ++d) vCa[r][d] += c[r] * crL[wave][h * 8 + r][i][d];
      }
      // finalize vector outputs; merge i-parity partials across lane^8
      float outv[8][3];
#pragma unroll
      for (int r = 0; r < 8; ++r) {
        int m = h * 8 + r;
        float ss = shsL[wave][m];
#pragma unroll
        for (int d = 0; d < 3; ++d)
          outv[r][d] = vAa[r] * shvL[wave][m][d] + vBa[r][d] * ss + vCa[r][d];
      }
#pragma unroll
      for (int r = 0; r < 8; ++r)
#pragma unroll
        for (int d = 0; d < 3; ++d) outv[r][d] += __shfl_xor(outv[r][d], 8);

      if (!isK) {
#pragma unroll
        for (int r = 0; r < 8; ++r)
          vout[(size_t)(e0 + h * 8 + r) * DD + nn] = souts[r];
        if (nn < 8)
#pragma unroll
          for (int r = 0; r < 8; ++r)
#pragma unroll
            for (int d = 0; d < 3; ++d)
              vout[(size_t)(e0 + h * 8 + r) * DD + NS + nn * 3 + d] = outv[r][d];
      } else {
        float part[8];
#pragma unroll
        for (int r = 0; r < 8; ++r) {
          int m = h * 8 + r;
          int s = srcL[wave][m];
          float p = souts[r] * q[(size_t)s * DD + nn];
          if (nn < 8) {
#pragma unroll
            for (int d = 0; d < 3; ++d)
              p += outv[r][d] * q[(size_t)s * DD + NS + nn * 3 + d];
          }
          part[r] = p;
        }
#pragma unroll
        for (int r = 0; r < 8; ++r) {
          float p = part[r];
          p += __shfl_xor(p, 1); p += __shfl_xor(p, 2);
          p += __shfl_xor(p, 4); p += __shfl_xor(p, 8);
          part[r] = p;
        }
        if (nn == 0) {
#pragma unroll
          for (int r = 0; r < 8; ++r) {
            int m = h * 8 + r;
            attn[e0 + m] = part[r];
            atomicMax(&mEnc[srcL[wave][m]], encF(part[r]));
          }
        }
      }
    }
  }
}

// ---------------------------------------------------------------------------
// Softmax / scatter / batchnorm helpers
// ---------------------------------------------------------------------------
__global__ __launch_bounds__(256)
void decode_m_kernel(const unsigned int* __restrict__ mEnc, float* __restrict__ mF) {
  int n = blockIdx.x * 256 + threadIdx.x;
  if (n >= N_NODES) return;
  unsigned int u = mEnc[n];
  float m = decF(u);
  if (u == 0u || !isfinite(m)) m = 0.f;
  mF[n] = m;
}

__global__ __launch_bounds__(256)
void exp_denom_kernel(const float* __restrict__ attn, const int* __restrict__ eidx,
                      const float* __restrict__ mF, float* __restrict__ eBuf,
                      float* __restrict__ denom) {
  int e = blockIdx.x * 256 + threadIdx.x;
  if (e >= N_EDGES) return;
  int s = eidx[N_EDGES + e];
  float ev = expf(attn[e] - mF[s]);
  eBuf[e] = ev;
  atomicAdd(&denom[s], ev);
}

__global__ __launch_bounds__(256)
void scatter_upd_kernel(const float* __restrict__ vout, const float* __restrict__ eBuf,
                        const float* __restrict__ denom, const int* __restrict__ eidx,
                        float* __restrict__ upd) {
  int i = blockIdx.x * 256 + threadIdx.x;
  if (i >= N_EDGES * DD) return;
  int e = i / DD;
  int c = i - e * DD;
  int s = eidx[N_EDGES + e];
  float a = eBuf[e] / denom[s];
  atomicAdd(&upd[(size_t)s * DD + c], vout[i] * a);
}

__global__ __launch_bounds__(256)
void stats_kernel(const float* __restrict__ atom, const float* __restrict__ upd,
                  float* __restrict__ stats) {
  __shared__ float part[40];
  int tid = threadIdx.x;
  if (tid < 40) part[tid] = 0.f;
  __syncthreads();
  int n = blockIdx.x * 256 + tid;
  if (n < N_NODES) {
#pragma unroll
    for (int c = 0; c < NS; ++c) {
      float x = atom[(size_t)n * DD + c] + upd[(size_t)n * DD + c];
      atomicAdd(&part[c], x);
      atomicAdd(&part[16 + c], x * x);
    }
#pragma unroll
    for (int o = 0; o < NV; ++o) {
      float vv = 0.f;
#pragma unroll
      for (int d = 0; d < 3; ++d) {
        float x = atom[(size_t)n * DD + NS + o * 3 + d] + upd[(size_t)n * DD + NS + o * 3 + d];
        vv += x * x;
      }
      atomicAdd(&part[32 + o], vv * (1.0f / 3.0f));
    }
  }
  __syncthreads();
  if (tid < 40) atomicAdd(&stats[tid], part[tid]);
}

__global__ __launch_bounds__(256)
void apply_bn_kernel(const float* __restrict__ atom, const float* __restrict__ upd,
                     const float* __restrict__ stats, const float* __restrict__ bnws,
                     const float* __restrict__ bnbs, const float* __restrict__ bnwv,
                     float* __restrict__ out) {
  int n = blockIdx.x * 256 + threadIdx.x;
  if (n >= N_NODES) return;
  const float invN = 1.0f / (float)N_NODES;
#pragma unroll
  for (int c = 0; c < NS; ++c) {
    float mu  = stats[c] * invN;
    float var = stats[16 + c] * invN - mu * mu;
    float x = atom[(size_t)n * DD + c] + upd[(size_t)n * DD + c];
    out[(size_t)n * DD + c] = (x - mu) / sqrtf(var + EPSBN) * bnws[c] + bnbs[c];
  }
#pragma unroll
  for (int o = 0; o < NV; ++o) {
    float norm = stats[32 + o] * invN;
    float sc = bnwv[o] / sqrtf(norm + EPSBN);
#pragma unroll
    for (int d = 0; d < 3; ++d) {
      float x = atom[(size_t)n * DD + NS + o * 3 + d] + upd[(size_t)n * DD + NS + o * 3 + d];
      out[(size_t)n * DD + NS + o * 3 + d] = x * sc;
    }
  }
}

// ---------------------------------------------------------------------------
extern "C" void kernel_launch(void* const* d_in, const int* in_sizes, int n_in,
                              void* d_out, int out_size, void* d_ws, size_t ws_size,
                              hipStream_t stream) {
  (void)in_sizes; (void)n_in; (void)out_size; (void)ws_size;
  const float* atom = (const float*)d_in[0];
  const float* ef   = (const float*)d_in[1];
  const float* esh  = (const float*)d_in[2];
  const float* Wqs  = (const float*)d_in[3];
  const float* Wqv  = (const float*)d_in[4];
  const float* kw1  = (const float*)d_in[5];
  const float* kb1  = (const float*)d_in[6];
  const float* kw2  = (const float*)d_in[7];
  const float* kb2  = (const float*)d_in[8];
  const float* vw1  = (const float*)d_in[9];
  const float* vb1  = (const float*)d_in[10];
  const float* vw2  = (const float*)d_in[11];
  const float* vb2  = (const float*)d_in[12];
  const float* bnws = (const float*)d_in[13];
  const float* bnbs = (const float*)d_in[14];
  const float* bnwv = (const float*)d_in[15];
  const int*   eidx = (const int*)d_in[16];

  float* ws = (float*)d_ws;            // workspace layout (floats), ~30.2 MB total
  float* q     = ws;                   // 400000
  float* attn  = ws + 400000;          // 160000
  float* eBuf  = ws + 560000;          // 160000
  float* mF    = ws + 720000;          // 10000
  float* denom = ws + 730000;          // 10000
  float* upd   = ws + 740000;          // 400000
  float* stats = ws + 1140000;         // 40
  unsigned int* mEnc = (unsigned int*)(ws + 1140064);   // 10000 u32
  float* vout  = ws + 1150064;         // 6,400,000

  hipMemsetAsync(denom, 0, N_NODES * sizeof(float), stream);
  hipMemsetAsync(upd,   0, (size_t)N_NODES * DD * sizeof(float), stream);
  hipMemsetAsync(stats, 0, 40 * sizeof(float), stream);
  hipMemsetAsync(mEnc,  0, N_NODES * sizeof(unsigned int), stream);

  node_q_kernel<<<(N_NODES + 255) / 256, 256, 0, stream>>>(atom, Wqs, Wqv, q);

  // k-pass: attention logits + segment max
  edge_pass_kernel<<<1250, BW * 32, 0, stream>>>(atom, ef, esh, kw1, kb1, kw2, kb2,
                                                 eidx, q, vout, attn, mEnc, 1);
  // v-pass: per-edge messages
  edge_pass_kernel<<<1250, BW * 32, 0, stream>>>(atom, ef, esh, vw1, vb1, vw2, vb2,
                                                 eidx, q, vout, attn, mEnc, 0);

  decode_m_kernel<<<(N_NODES + 255) / 256, 256, 0, stream>>>(mEnc, mF);
  exp_denom_kernel<<<(N_EDGES + 255) / 256, 256, 0, stream>>>(attn, eidx, mF, eBuf, denom);
  scatter_upd_kernel<<<(N_EDGES * DD + 255) / 256, 256, 0, stream>>>(vout, eBuf, denom, eidx, upd);
  stats_kernel<<<(N_NODES + 255) / 256, 256, 0, stream>>>(atom, upd, stats);
  apply_bn_kernel<<<(N_NODES + 255) / 256, 256, 0, stream>>>(atom, upd, stats, bnws, bnbs,
                                                             bnwv, (float*)d_out);
  hipMemcpyAsync((float*)d_out + (size_t)N_NODES * DD, ef,
                 (size_t)N_EDGES * HE * sizeof(float), hipMemcpyDeviceToDevice, stream);
}
